// AttentionVisit_47175920779497
// MI455X (gfx1250) — matrix operand order repaired and checked by reading, not compiled, not run on hardware
//
#include <hip/hip_runtime.h>
#include <hip/hip_bf16.h>

// Problem dims (fixed by the reference)
#define BB 128
#define SS 1024
#define DD 256
#define AA 256
#define OO 256

#define LDB 264   // padded bf16 row stride (ushort elements); 528B = 16B multiple
#define LDP 264   // padded f32 row stride (float elements);  1056B = 16B multiple

typedef __attribute__((ext_vector_type(16))) __bf16          v16bf;
typedef __attribute__((ext_vector_type(16))) unsigned short  v16u;
typedef __attribute__((ext_vector_type(8)))  unsigned short  v8u;
typedef __attribute__((ext_vector_type(8)))  float           v8f;

static __device__ inline unsigned short f2bf(float f) {
  unsigned int x = __float_as_uint(f);
  unsigned int r = x + 0x7FFFu + ((x >> 16) & 1u);   // RNE
  return (unsigned short)(r >> 16);
}

// A fragment: 16x32 bf16 tile from LDS, rows mb..mb+15, K cols kb..kb+31.
// ISA layout: lane l -> M = l&15 ; elem j: K = (j<8 ? g*8+j : 16+g*8+(j-8)), g=l>>4.
// Each lane = two contiguous 16B runs -> 2x ds_load_b128.
static __device__ inline v16u load_a_frag(const unsigned short* buf, int mb,
                                          int kb, int lane) {
  const unsigned short* row = buf + (mb + (lane & 15)) * LDB + kb + (lane >> 4) * 8;
  v8u lo = *(const v8u*)row;          // K = g*8 .. g*8+7
  v8u hi = *(const v8u*)(row + 16);   // K = 16+g*8 .. 16+g*8+7
  v16u f;
#pragma unroll
  for (int j = 0; j < 8; ++j) { f[j] = lo[j]; f[8 + j] = hi[j]; }
  return f;
}

// B fragment: 32x16 bf16 tile (K x N) from TRANSPOSED weights Wt[N][K] (ldk=256).
// ISA layout: lane l -> N = l&15 ; elem j: K = (j<8 ? g*8+j : 16+g*8+(j-8)), g=l>>4.
// Each lane = two 16B runs 32B apart -> 2x global_load_b128.
static __device__ inline v16u load_b_frag(const unsigned short* Wt, int kb,
                                          int nb, int lane) {
  const unsigned short* p = Wt + (size_t)(nb + (lane & 15)) * DD + kb + (lane >> 4) * 8;
  v8u lo = *(const v8u*)p;
  v8u hi = *(const v8u*)(p + 16);
  v16u f;
#pragma unroll
  for (int j = 0; j < 8; ++j) { f[j] = lo[j]; f[8 + j] = hi[j]; }
  return f;
}

static __device__ inline v8f wmma_bf16(v16u a, v16u b, v8f c) {
  union { v16u u; v16bf bf; } ua, ub;
  ua.u = a; ub.u = b;
  return __builtin_amdgcn_wmma_f32_16x16x32_bf16(
      false, ua.bf, false, ub.bf, (short)0, c, false, false);
}

// ---------------------------------------------------------------------------
// Kernel 0: convert + transpose W [D,A] -> Wt [A,D] bf16, u_o [A,O] -> Ut [O,A]
// ---------------------------------------------------------------------------
__global__ __launch_bounds__(256) void cvt_kernel(
    const float* __restrict__ w, const float* __restrict__ uo,
    unsigned short* __restrict__ Wt, unsigned short* __restrict__ Ut) {
  int idx = blockIdx.x * 256 + threadIdx.x;   // 65536 threads
  int r = idx >> 8;          // output row (a for Wt, o for Ut)
  int c = idx & 255;         // output col (d for Wt, a for Ut)
  Wt[idx] = f2bf(w[c * AA + r]);   // Wt[a][d] = W[d][a]
  Ut[idx] = f2bf(uo[c * OO + r]);  // Ut[o][a] = Uo[a][o]
}

// ---------------------------------------------------------------------------
// Kernel 1: fused tile kernel. One block (256 thr = 8 waves) per
// (batch, 32-row s-tile). t = x@W + b -> L2 normalize -> tanh -> vu dot,
// vu_o = v@Uo -> row softmax over O -> betas; masked vu logits -> ws.
// ---------------------------------------------------------------------------
__global__ __launch_bounds__(256) void attn_tile_kernel(
    const float* __restrict__ x, const float* __restrict__ mask,
    const unsigned short* __restrict__ Wt, const float* __restrict__ b_omega,
    const float* __restrict__ u_omega, const unsigned short* __restrict__ Ut,
    float* __restrict__ vu_ws, float* __restrict__ betas_out) {
  __shared__ __align__(16) unsigned short sb[32 * LDB];  // x tile bf16, then v tile bf16
  __shared__ __align__(16) float ts[32 * LDP];           // t (f32), then vu_o (f32)
  __shared__ float rs[32], vurow[32], mxs[32], invs[32];

  const int tid  = threadIdx.x;
  const int wave = tid >> 5;
  const int lane = tid & 31;
  const int b    = blockIdx.x >> 5;       // 32 s-tiles per batch
  const int s0   = (blockIdx.x & 31) * 32;
  const int nb0  = wave * 32;             // this wave's 32 output columns

  // --- stage x tile (32 x 256) as bf16: float4 loads, packed 8B LDS stores ---
  {
    const float4* x4 = (const float4*)(x + ((size_t)b * SS + s0) * DD);
    for (int idx = tid; idx < 32 * 64; idx += 256) {
      int r = idx >> 6, c4 = idx & 63;
      float4 v = x4[r * 64 + c4];
      uint2 p;
      p.x = (unsigned)f2bf(v.x) | ((unsigned)f2bf(v.y) << 16);
      p.y = (unsigned)f2bf(v.z) | ((unsigned)f2bf(v.w) << 16);
      *(uint2*)(sb + r * LDB + c4 * 4) = p;
    }
  }
  __syncthreads();

  // --- GEMM1: t = x @ W  (2 M-tiles x 2 N-tiles per wave, K = 8 x 32) ---
  v8f acc[2][2];
#pragma unroll
  for (int mt = 0; mt < 2; ++mt)
#pragma unroll
    for (int nt = 0; nt < 2; ++nt) acc[mt][nt] = (v8f){0,0,0,0,0,0,0,0};

  for (int kt = 0; kt < 8; ++kt) {
    int kb = kt * 32;
    v16u a0 = load_a_frag(sb,  0, kb, lane);
    v16u a1 = load_a_frag(sb, 16, kb, lane);
#pragma unroll
    for (int nt = 0; nt < 2; ++nt) {
      v16u bf = load_b_frag(Wt, kb, nb0 + nt * 16, lane);
      acc[0][nt] = wmma_bf16(a0, bf, acc[0][nt]);
      acc[1][nt] = wmma_bf16(a1, bf, acc[1][nt]);
    }
  }

  // --- stage t + bias into LDS f32 ---
#pragma unroll
  for (int mt = 0; mt < 2; ++mt)
#pragma unroll
    for (int nt = 0; nt < 2; ++nt) {
      int col = nb0 + nt * 16 + (lane & 15);
      float bo = b_omega[col];
      int rbase = mt * 16 + (lane >> 4) * 8;
#pragma unroll
      for (int r = 0; r < 8; ++r)
        ts[(rbase + r) * LDP + col] = acc[mt][nt][r] + bo;
    }
  __syncthreads();

  // --- per-row inverse L2 norm over A=256 ---
  if (tid < 32) {
    float s = 0.f;
    for (int c = 0; c < 256; ++c) { float v = ts[tid * LDP + c]; s += v * v; }
    rs[tid] = 1.0f / fmaxf(sqrtf(s), 1e-12f);
  }
  __syncthreads();

  // --- v = tanh(t * invnorm): f32 back into ts, bf16 into sb ---
  for (int idx = tid; idx < 32 * 256; idx += 256) {
    int r = idx >> 8, c = idx & 255;
    float v = tanhf(ts[r * LDP + c] * rs[r]);
    ts[r * LDP + c] = v;
    sb[r * LDB + c] = f2bf(v);
  }
  __syncthreads();

  // --- vu = v . u_omega (per row, f32) ---
  if (tid < 32) {
    float s = 0.f;
    for (int c = 0; c < 256; ++c) s += ts[tid * LDP + c] * u_omega[c];
    vurow[tid] = s;
  }

  // --- GEMM2: vu_o = v @ u_omega_o ---
  v8f acc2[2][2];
#pragma unroll
  for (int mt = 0; mt < 2; ++mt)
#pragma unroll
    for (int nt = 0; nt < 2; ++nt) acc2[mt][nt] = (v8f){0,0,0,0,0,0,0,0};

  for (int kt = 0; kt < 8; ++kt) {
    int kb = kt * 32;
    v16u a0 = load_a_frag(sb,  0, kb, lane);
    v16u a1 = load_a_frag(sb, 16, kb, lane);
#pragma unroll
    for (int nt = 0; nt < 2; ++nt) {
      v16u bf = load_b_frag(Ut, kb, nb0 + nt * 16, lane);
      acc2[0][nt] = wmma_bf16(a0, bf, acc2[0][nt]);
      acc2[1][nt] = wmma_bf16(a1, bf, acc2[1][nt]);
    }
  }
  __syncthreads();   // vu (reads ts) done before ts is overwritten

  // --- stage vu_o into ts ---
#pragma unroll
  for (int mt = 0; mt < 2; ++mt)
#pragma unroll
    for (int nt = 0; nt < 2; ++nt) {
      int col = nb0 + nt * 16 + (lane & 15);
      int rbase = mt * 16 + (lane >> 4) * 8;
#pragma unroll
      for (int r = 0; r < 8; ++r)
        ts[(rbase + r) * LDP + col] = acc2[mt][nt][r];
    }
  __syncthreads();

  // --- per-row softmax stats over O=256; emit masked vu logit ---
  if (tid < 32) {
    float mx = -3.4e38f;
    for (int c = 0; c < 256; ++c) mx = fmaxf(mx, ts[tid * LDP + c]);
    float sm = 0.f;
    for (int c = 0; c < 256; ++c) sm += expf(ts[tid * LDP + c] - mx);
    mxs[tid] = mx;
    invs[tid] = 1.0f / sm;   // sm >= 1 (exp(0) term present)
    size_t si = (size_t)b * SS + s0 + tid;
    vu_ws[si] = vurow[tid] * mask[si];
  }
  __syncthreads();

  // --- betas out: float4 stores (zeros where mask==0, per masked_softmax) ---
  for (int idx = tid; idx < 32 * 64; idx += 256) {
    int r = idx >> 6, c = (idx & 63) * 4;
    size_t si = (size_t)b * SS + s0 + r;
    float mk = mask[si];
    float4 o;
    if (mk != 0.f) {
      float4 t4 = *(const float4*)(ts + r * LDP + c);
      float m = mxs[r], iv = invs[r];
      o.x = expf(t4.x - m) * iv;
      o.y = expf(t4.y - m) * iv;
      o.z = expf(t4.z - m) * iv;
      o.w = expf(t4.w - m) * iv;
    } else {
      o = (float4){0.f, 0.f, 0.f, 0.f};
    }
    *(float4*)(betas_out + si * OO + c) = o;
  }
}

// ---------------------------------------------------------------------------
// Kernel 2: alphas = masked_softmax(vu*mask) over S=1024, one block per batch
// ---------------------------------------------------------------------------
__global__ __launch_bounds__(256) void alphas_kernel(
    const float* __restrict__ vu_ws, const float* __restrict__ mask,
    float* __restrict__ alphas) {
  __shared__ float red[256];
  int b = blockIdx.x, tid = threadIdx.x;
  float v[4], m[4];
  float mx = -3.4e38f;
#pragma unroll
  for (int i = 0; i < 4; ++i) {
    size_t si = (size_t)b * SS + tid + i * 256;
    v[i] = vu_ws[si];            // already vu*mask
    m[i] = mask[si];
    mx = fmaxf(mx, v[i]);
  }
  red[tid] = mx; __syncthreads();
  for (int off = 128; off; off >>= 1) {
    if (tid < off) red[tid] = fmaxf(red[tid], red[tid + off]);
    __syncthreads();
  }
  mx = red[0]; __syncthreads();
  float e[4], sm = 0.f;
#pragma unroll
  for (int i = 0; i < 4; ++i) { e[i] = expf(v[i] - mx) * m[i]; sm += e[i]; }
  red[tid] = sm; __syncthreads();
  for (int off = 128; off; off >>= 1) {
    if (tid < off) red[tid] += red[tid + off];
    __syncthreads();
  }
  sm = red[0];
  float denom = (sm == 0.f) ? 1.f : sm;
#pragma unroll
  for (int i = 0; i < 4; ++i)
    alphas[(size_t)b * SS + tid + i * 256] = e[i] / denom;
}

// ---------------------------------------------------------------------------
// Kernel 3: output[b,o] = sum_s x[b,s,o] * alphas[b,s] * betas[b,s,o]
// ---------------------------------------------------------------------------
__global__ __launch_bounds__(256) void out_kernel(
    const float* __restrict__ x, const float* __restrict__ alphas,
    const float* __restrict__ betas, float* __restrict__ out) {
  int b = blockIdx.x, o = threadIdx.x;
  float acc = 0.f;
  for (int s = 0; s < SS; ++s) {
    float a = alphas[(size_t)b * SS + s];
    size_t base = ((size_t)b * SS + s) * OO + o;
    acc += x[base] * a * betas[base];
  }
  out[(size_t)b * OO + o] = acc;
}

// ---------------------------------------------------------------------------
extern "C" void kernel_launch(void* const* d_in, const int* in_sizes, int n_in,
                              void* d_out, int out_size, void* d_ws, size_t ws_size,
                              hipStream_t stream) {
  const float* x    = (const float*)d_in[0];   // [B,S,D]
  const float* mask = (const float*)d_in[1];   // [B,S]
  const float* w    = (const float*)d_in[2];   // [D,A]
  const float* bo   = (const float*)d_in[3];   // [A]
  const float* u    = (const float*)d_in[4];   // [A]
  const float* uo   = (const float*)d_in[5];   // [A,O]

  float* out    = (float*)d_out;               // [B,O]
  float* alphas = out + (size_t)BB * OO;       // [B,S]
  float* betas  = alphas + (size_t)BB * SS;    // [B,S,O]

  unsigned short* Wt = (unsigned short*)d_ws;          // [A,D] bf16 (transposed)
  unsigned short* Ut = Wt + (size_t)DD * AA;           // [O,A] bf16 (transposed)
  float* vu_ws = (float*)(Ut + (size_t)AA * OO);       // [B,S] masked logits

  cvt_kernel<<<(DD * AA) / 256, 256, 0, stream>>>(w, uo, Wt, Ut);
  attn_tile_kernel<<<BB * (SS / 32), 256, 0, stream>>>(x, mask, Wt, bo, u, Ut,
                                                       vu_ws, betas);
  alphas_kernel<<<BB, 256, 0, stream>>>(vu_ws, mask, alphas);
  out_kernel<<<BB, 256, 0, stream>>>(x, alphas, betas, out);
}
